// MambaTemporalBlock_22076131902162
// MI455X (gfx1250) — matrix operand order repaired and checked by reading, not compiled, not run on hardware
//
#include <hip/hip_runtime.h>
#include <cstring>

// ---------------- problem constants ----------------
#define BN   4096
#define T    64
#define D    128
#define DI   256
#define DS   16
#define DR   8
#define DC   4

// LDS strides (elements). Padded where WMMA A-fragments read rows.
#define XN_STR   136   // 128 + 8 pad (bf16)
#define XB_STR   256   // conv reads per-channel, no pad needed
#define SZ_STR   256
#define XC_STR   264   // 256 + 8 pad (A-matrix for xproj)
#define Y_STR    264   // 256 + 8 pad (A-matrix for out GEMM)
#define DBL_STR  48    // 40 cols padded to 48 (fp32)

#define SMEM_BYTES ((T*XN_STR + T*XB_STR + T*SZ_STR + T*XC_STR + T*Y_STR)*2 + T*DBL_STR*4)

typedef __attribute__((ext_vector_type(16))) __bf16 bf16x16;
typedef __attribute__((ext_vector_type(8)))  float  f32x8;

union FragU { bf16x16 v; uint4 q[2]; };

__device__ __forceinline__ __bf16 f2bf(float x) {
    union { float f; unsigned u; } a; a.f = x;
    unsigned r = (a.u + 0x7FFFu + ((a.u >> 16) & 1u)) >> 16;
    union { unsigned short s; __bf16 b; } o; o.s = (unsigned short)r;
    return o.b;
}
__device__ __forceinline__ float bf2f(__bf16 b) {
    union { __bf16 b; unsigned short s; } i; i.b = b;
    union { unsigned u; float f; } o; o.u = ((unsigned)i.s) << 16;
    return o.f;
}
// fast silu: x * v_rcp_f32(1+exp(-x))  (avoid IEEE div ladder)
__device__ __forceinline__ float siluf(float x) {
    return x * __builtin_amdgcn_rcpf(1.f + __expf(-x));
}

// A fragment (16x32 bf16 tile, row-major in LDS, row stride strideE elements)
__device__ __forceinline__ bf16x16 load_frag_A(const __bf16* tile, int strideE, int K0, int lane) {
    int m = lane & 15, half = lane >> 4;
    const __bf16* p = tile + m * strideE + K0 + 8 * half;
    FragU f;
    f.q[0] = *(const uint4*)(p);
    f.q[1] = *(const uint4*)(p + 16);
    return f.v;
}

// B fragment (32x16): B[k][n] = W[n][k], W row-major with row stride Krow.
// Lane l: n = l&15, needs k at K0 + 8*(l>>4) + [0..7] and K0 + 16 + 8*(l>>4) + [0..7].
__device__ __forceinline__ bf16x16 load_frag_B(const __bf16* W, int Krow, int N0, int K0, int lane) {
    int n = lane & 15, kb = (lane >> 4) * 8;
    const __bf16* p = W + (size_t)(N0 + n) * Krow + K0 + kb;
    FragU f;
    f.q[0] = *(const uint4*)(p);
    f.q[1] = *(const uint4*)(p + 16);
    return f.v;
}

__device__ __forceinline__ f32x8 wmma_bf16(bf16x16 a, bf16x16 b, f32x8 c) {
    return __builtin_amdgcn_wmma_f32_16x16x32_bf16(false, a, false, b, (short)0, c, false, false);
}

// ---------------- prep: fp32 weights -> bf16 (L2-resident copies) ----------------
__global__ void prep_weights(const float* __restrict__ Win,   // (512,128)
                             const float* __restrict__ Wxp,   // (40,256)
                             const float* __restrict__ Wout,  // (128,256)
                             __bf16* __restrict__ Winb,
                             __bf16* __restrict__ Wxpb,       // zero-padded to (48,256)
                             __bf16* __restrict__ Woutb) {
    int i = blockIdx.x * blockDim.x + threadIdx.x;
    if (i < 2 * DI * D) Winb[i] = f2bf(Win[i]);
    if (i < 48 * DI) {
        int r = i / DI, c = i % DI;
        Wxpb[i] = (r < DR + 2 * DS) ? f2bf(Wxp[r * DI + c]) : f2bf(0.f);
    }
    if (i < D * DI) Woutb[i] = f2bf(Wout[i]);
}

// ---------------- fused Mamba block: one workgroup per batch ----------------
__global__ __launch_bounds__(256) void mamba_fused(
    const float* __restrict__ x,        // (BN,T,D)
    const float* __restrict__ ln_g, const float* __restrict__ ln_b,
    const float* __restrict__ conv_w,   // (DI,DC)
    const float* __restrict__ conv_b,
    const float* __restrict__ W_dt,     // (DI,DR)
    const float* __restrict__ b_dt,
    const float* __restrict__ A_log,    // (DI,DS)
    const float* __restrict__ D_param,
    const __bf16* __restrict__ Winb,    // (512,128) bf16
    const __bf16* __restrict__ Wxpb,    // (48,256)  bf16
    const __bf16* __restrict__ Woutb,   // (128,256) bf16
    float* __restrict__ out)            // (BN,T,D)
{
    extern __shared__ char smem[];
    __bf16* s_xn  = (__bf16*)smem;                 // T x XN_STR
    __bf16* s_xb  = s_xn + T * XN_STR;             // T x XB_STR
    __bf16* s_sz  = s_xb + T * XB_STR;             // T x SZ_STR  (silu(z))
    __bf16* s_xc  = s_sz + T * SZ_STR;             // T x XC_STR
    __bf16* s_y   = s_xc + T * XC_STR;             // T x Y_STR
    float*  s_dbl = (float*)(s_y + T * Y_STR);     // T x DBL_STR fp32

    const int tid  = threadIdx.x;
    const int lane = tid & 31;
    const int wave = tid >> 5;
    const size_t bo = (size_t)blockIdx.x * T * D;
    const float* xg = x + bo;
    float* outg = out + bo;

    // ---- Stage 0: LayerNorm -> s_xn (bf16). 4 threads per row, 32 elems each.
    {
        const int row = tid >> 2, q = tid & 3;
        const float* xr = xg + row * D + q * 32;
        float v[32];
        #pragma unroll
        for (int i = 0; i < 8; ++i) {
            float4 f = ((const float4*)xr)[i];
            v[4*i+0] = f.x; v[4*i+1] = f.y; v[4*i+2] = f.z; v[4*i+3] = f.w;
        }
        float sm = 0.f, ss = 0.f;
        #pragma unroll
        for (int i = 0; i < 32; ++i) { sm += v[i]; ss += v[i] * v[i]; }
        #pragma unroll
        for (int m = 1; m < 4; m <<= 1) {
            sm += __shfl_xor(sm, m, 32);
            ss += __shfl_xor(ss, m, 32);
        }
        float mean = sm * (1.f / D);
        float var  = ss * (1.f / D) - mean * mean;
        float rstd = rsqrtf(var + 1e-5f);
        __bf16* dst = s_xn + row * XN_STR + q * 32;
        #pragma unroll
        for (int i = 0; i < 32; ++i) {
            float g = ln_g[q * 32 + i], bb = ln_b[q * 32 + i];
            dst[i] = f2bf((v[i] - mean) * rstd * g + bb);
        }
    }
    __syncthreads();

    // ---- Stage 1: GEMM1  xz = xn @ W_in^T   (M=64, N=512, K=128)
    // wave w handles n-tiles 4w..4w+3; waves 0-3 -> xb, waves 4-7 -> silu(z)
    // The xb/z split is wave-uniform -> scalar branch, no exec-mask churn.
    {
        const int n  = lane & 15, half = lane >> 4;
        for (int ntl = 0; ntl < 4; ++ntl) {
            const int N0 = (wave * 4 + ntl) * 16;
            bf16x16 bfr[4];
            #pragma unroll
            for (int kt = 0; kt < 4; ++kt) bfr[kt] = load_frag_B(Winb, D, N0, kt * 32, lane);
            for (int mt = 0; mt < 4; ++mt) {
                f32x8 acc = {0.f,0.f,0.f,0.f,0.f,0.f,0.f,0.f};
                #pragma unroll
                for (int kt = 0; kt < 4; ++kt) {
                    bf16x16 a = load_frag_A(s_xn + mt * 16 * XN_STR, XN_STR, kt * 32, lane);
                    acc = wmma_bf16(a, bfr[kt], acc);
                }
                if (wave < 4) {            // cols 0..255 -> xb
                    const int col = N0 + n;
                    #pragma unroll
                    for (int r = 0; r < 8; ++r)
                        s_xb[(mt * 16 + r + 8 * half) * XB_STR + col] = f2bf(acc[r]);
                } else {                   // cols 256..511 -> silu(z)
                    const int col = N0 + n - DI;
                    #pragma unroll
                    for (int r = 0; r < 8; ++r)
                        s_sz[(mt * 16 + r + 8 * half) * SZ_STR + col] = f2bf(siluf(acc[r]));
                }
            }
        }
    }
    __syncthreads();

    // ---- Stage 2: depthwise causal conv (DC=4) + SiLU -> s_xc. Thread = channel.
    {
        const int d = tid;
        float cw[DC];
        #pragma unroll
        for (int j = 0; j < DC; ++j) cw[j] = conv_w[d * DC + j];
        const float cb = conv_b[d];
        for (int t = 0; t < T; ++t) {
            float acc = cb;
            #pragma unroll
            for (int j = 0; j < DC; ++j) {
                int tj = t - (DC - 1) + j;
                if (tj >= 0) acc += bf2f(s_xb[tj * XB_STR + d]) * cw[j];
            }
            s_xc[t * XC_STR + d] = f2bf(siluf(acc));
        }
    }
    __syncthreads();

    // ---- Stage 3: xproj GEMM  dbl = xc @ W_xproj^T  (M=64, N=40->48, K=256)
    {
        const int n = lane & 15, half = lane >> 4;
        for (int j = wave; j < 12; j += 8) {
            const int mt = j / 3, N0 = (j % 3) * 16;
            f32x8 acc = {0.f,0.f,0.f,0.f,0.f,0.f,0.f,0.f};
            #pragma unroll
            for (int kt = 0; kt < 8; ++kt) {
                bf16x16 a = load_frag_A(s_xc + mt * 16 * XC_STR, XC_STR, kt * 32, lane);
                bf16x16 b = load_frag_B(Wxpb, DI, N0, kt * 32, lane);
                acc = wmma_bf16(a, b, acc);
            }
            const int col = N0 + n;
            if (col < DR + 2 * DS) {
                #pragma unroll
                for (int r = 0; r < 8; ++r)
                    s_dbl[(mt * 16 + r + 8 * half) * DBL_STR + col] = acc[r];
            }
        }
    }
    __syncthreads();

    // ---- Stage 4+5: dt projection + softplus + sequential SSM scan + gating.
    // Thread = channel d; h[16] in registers; dbl row broadcast from LDS.
    {
        const int d = tid;
        float wdt[DR];
        #pragma unroll
        for (int r = 0; r < DR; ++r) wdt[r] = W_dt[d * DR + r];
        const float bdt = b_dt[d];
        const float Dp  = D_param[d];
        float Aq[DS];
        #pragma unroll
        for (int s = 0; s < DS; ++s) Aq[s] = -__expf(A_log[d * DS + s]);
        float h[DS];
        #pragma unroll
        for (int s = 0; s < DS; ++s) h[s] = 0.f;

        for (int t = 0; t < T; ++t) {
            const float* db = s_dbl + t * DBL_STR;
            float acc = bdt;
            #pragma unroll
            for (int r = 0; r < DR; ++r) acc += db[r] * wdt[r];
            // softplus via fast exp/log (guarded for large x)
            float dtv = (acc > 20.f) ? acc : __logf(1.f + __expf(acc));
            float xt  = bf2f(s_xc[t * XC_STR + d]);
            float dtx = dtv * xt;
            float y = 0.f;
            #pragma unroll
            for (int s = 0; s < DS; ++s) {
                float dA = __expf(dtv * Aq[s]);
                h[s] = h[s] * dA + dtx * db[DR + s];
                y += h[s] * db[DR + DS + s];
            }
            float yo = (y + xt * Dp) * bf2f(s_sz[t * SZ_STR + d]);
            s_y[t * Y_STR + d] = f2bf(yo);
        }
    }
    __syncthreads();

    // ---- Stage 6: out GEMM  out = y @ W_out^T + residual  (M=64, N=128, K=256)
    {
        const int n = lane & 15, half = lane >> 4;
        const int N0 = wave * 16;
        bf16x16 bfr[8];
        #pragma unroll
        for (int kt = 0; kt < 8; ++kt) bfr[kt] = load_frag_B(Woutb, DI, N0, kt * 32, lane);
        for (int mt = 0; mt < 4; ++mt) {
            f32x8 acc = {0.f,0.f,0.f,0.f,0.f,0.f,0.f,0.f};
            #pragma unroll
            for (int kt = 0; kt < 8; ++kt) {
                bf16x16 a = load_frag_A(s_y + mt * 16 * Y_STR, Y_STR, kt * 32, lane);
                acc = wmma_bf16(a, bfr[kt], acc);
            }
            const int col = N0 + n;
            #pragma unroll
            for (int r = 0; r < 8; ++r) {
                int row = mt * 16 + r + 8 * half;
                outg[row * D + col] = acc[r] + xg[row * D + col];
            }
        }
    }
}

extern "C" void kernel_launch(void* const* d_in, const int* in_sizes, int n_in,
                              void* d_out, int out_size, void* d_ws, size_t ws_size,
                              hipStream_t stream) {
    const float* x      = (const float*)d_in[0];
    const float* ln_g   = (const float*)d_in[1];
    const float* ln_b   = (const float*)d_in[2];
    const float* W_in   = (const float*)d_in[3];
    const float* conv_w = (const float*)d_in[4];
    const float* conv_b = (const float*)d_in[5];
    const float* W_xproj= (const float*)d_in[6];
    const float* W_dt   = (const float*)d_in[7];
    const float* b_dt   = (const float*)d_in[8];
    const float* A_log  = (const float*)d_in[9];
    const float* D_par  = (const float*)d_in[10];
    const float* W_out  = (const float*)d_in[11];
    float* out = (float*)d_out;

    // workspace: bf16 weight copies (all 16B-aligned)
    __bf16* Winb  = (__bf16*)d_ws;                       // 512*128
    __bf16* Wxpb  = Winb + 2 * DI * D;                   // 48*256 (padded)
    __bf16* Woutb = Wxpb + 48 * DI;                      // 128*256

    prep_weights<<<(2 * DI * D + 255) / 256, 256, 0, stream>>>(
        W_in, W_xproj, W_out, Winb, Wxpb, Woutb);

    hipFuncSetAttribute((const void*)mamba_fused,
                        hipFuncAttributeMaxDynamicSharedMemorySize, SMEM_BYTES);

    mamba_fused<<<BN, 256, SMEM_BYTES, stream>>>(
        x, ln_g, ln_b, conv_w, conv_b, W_dt, b_dt, A_log, D_par,
        Winb, Wxpb, Woutb, out);
}
